// LorentzMHA_14224931684390
// MI455X (gfx1250) — matrix operand order repaired and checked by reading, not compiled, not run on hardware
//
#include <hip/hip_runtime.h>
#include <hip/hip_bf16.h>

typedef __attribute__((ext_vector_type(16))) _Float16 v16h;
typedef __attribute__((ext_vector_type(8)))  _Float16 v8h;
typedef __attribute__((ext_vector_type(4)))  _Float16 v4h;
typedef __attribute__((ext_vector_type(8)))  float    v8f;

#define TOKENS 4096
#define DMODEL 128
#define HEADS  8
#define NSEQ   1024
#define OUTD   1017   // 8*127 + 1

// Build a 16x32 f16 A-fragment from a row-major row of halves.
// Layout: lane = m + 16*g; halves[0..15] = row[kt*32 + g*16 ..] (contiguous k).
static __device__ __forceinline__ v16h make_a_frag(const _Float16* row, int kt, int g) {
  v8h lo = *(const v8h*)(row + kt * 32 + g * 16);
  v8h hi = *(const v8h*)(row + kt * 32 + g * 16 + 8);
  v16h a;
#pragma unroll
  for (int i = 0; i < 8; ++i) { a[i] = lo[i]; a[8 + i] = hi[i]; }
  return a;
}

// ---------------- Kernel 0a: convert x fp32 -> f16 (once) ----------------
__global__ __launch_bounds__(256) void lmha_cvt_x_kernel(
    const float* __restrict__ x, _Float16* __restrict__ Xh)
{
  const int i = blockIdx.x * blockDim.x + threadIdx.x;   // 131072 = 4096*128/4
  const float4 f = ((const float4*)x)[i];
  v4h hv = {(_Float16)f.x, (_Float16)f.y, (_Float16)f.z, (_Float16)f.w};
  ((v4h*)Xh)[i] = hv;
}

// ------- Kernel 0b: convert W fp32 -> padded f16 Wh[3][8][128][128] -------
// Row e==127 is zero so projection GEMM needs no masking.
__global__ __launch_bounds__(256) void lmha_cvt_w_kernel(
    const float* __restrict__ Wq, const float* __restrict__ Wk,
    const float* __restrict__ Wv, _Float16* __restrict__ Wh)
{
  const int i   = blockIdx.x * blockDim.x + threadIdx.x;  // 98304 = 3*8*128*128/4
  const int d4  = i & 31;          // 32 float4 groups per 128-wide row
  const int row = i >> 5;          // mat*1024 + h*128 + e
  const int e   = row & 127;
  const int h   = (row >> 7) & 7;
  const int mat = row >> 10;
  const float* W = (mat == 0) ? Wq : (mat == 1) ? Wk : Wv;
  v4h hv = {(_Float16)0.f, (_Float16)0.f, (_Float16)0.f, (_Float16)0.f};
  if (e < 127) {
    const float4 f = ((const float4*)(W + ((size_t)h * 127 + e) * DMODEL))[d4];
    hv = (v4h){(_Float16)f.x, (_Float16)f.y, (_Float16)f.z, (_Float16)f.w};
  }
  ((v4h*)Wh)[i] = hv;
}

// ---------------- Kernel 1: Lorentz FC projections (Q, K, V) ----------------
// 6144 wave-tasks: 3 mats x 8 heads x 256 row-tiles of 16 tokens.
__global__ __launch_bounds__(128) void lmha_proj_kernel(
    const _Float16* __restrict__ Xh, const _Float16* __restrict__ Wh,
    const float* __restrict__ bq, const float* __restrict__ bk,
    const float* __restrict__ bv,
    _Float16* __restrict__ Qws, _Float16* __restrict__ Kws,
    _Float16* __restrict__ Vt)
{
  __shared__ float tile[4][16][132];   // per-wave 16x128 fp32 staging (+pad)
  __shared__ float trow[4][16];        // per-wave time coords (V path)

  const int wid = threadIdx.x >> 5;
  const int L   = threadIdx.x & 31;
  const int l16 = L & 15;
  const int g   = L >> 4;

  const int task    = blockIdx.x * 4 + wid;   // 0..6143
  const int mat     = task / 2048;            // 0=Q 1=K 2=V
  const int rem     = task % 2048;
  const int h       = rem / 256;
  const int rowbase = (rem % 256) * 16;       // token tile (flat over B*N)

  const float* bias = (mat == 0) ? bq : (mat == 1) ? bk : bv;
  const _Float16* Whh = Wh + ((size_t)(mat * HEADS + h)) * DMODEL * DMODEL;

  // A fragments: 16 token rows of Xh (pure f16 loads)
  v16h a[4];
  {
    const _Float16* row = Xh + (size_t)(rowbase + l16) * DMODEL;
#pragma unroll
    for (int kt = 0; kt < 4; ++kt) a[kt] = make_a_frag(row, kt, g);
  }

  // S = X @ W^T over 8 output tiles of 16 (row 127 of Wh is zero-padded)
#pragma unroll
  for (int nt = 0; nt < 8; ++nt) {
    const int e = nt * 16 + l16;               // B-fragment column per lane
    v8f acc = {0.f, 0.f, 0.f, 0.f, 0.f, 0.f, 0.f, 0.f};
#pragma unroll
    for (int kt = 0; kt < 4; ++kt) {
      v16h bf = *(const v16h*)(Whh + (size_t)e * DMODEL + kt * 32 + g * 16);
      acc = __builtin_amdgcn_wmma_f32_16x16x32_f16(false, a[kt], false, bf,
                                                   (short)0, acc, false, false);
    }
    const float bvad = (e < 127) ? bias[(size_t)h * 127 + e] : 0.0f;
#pragma unroll
    for (int r = 0; r < 8; ++r)
      tile[wid][r + 8 * g][nt * 16 + l16] = acc[r] + bvad;  // C layout -> LDS
  }
  __syncthreads();

  // Epilogue: hyperboloid lift t = sqrt(1 + ||s||^2)  (col 127 is 0)
  {
    const int row = L >> 1, hs = L & 1;
    float ss = 0.f;
#pragma unroll
    for (int j = 0; j < 64; ++j) { float v = tile[wid][row][hs * 64 + j]; ss += v * v; }
    ss += __shfl_xor(ss, 1, 32);
    const float t = sqrtf(1.0f + ss);

    if (mat < 2) {
      // row-major f16: [0]=(+-)t, [1+e]=space[e]; Q stores -t (Minkowski sign)
      _Float16* outp = ((mat == 0) ? Qws : Kws) +
                       ((size_t)h * TOKENS + rowbase + row) * DMODEL + hs * 64;
      const float sgn = (mat == 0) ? -1.0f : 1.0f;
      _Float16 buf[64];
#pragma unroll
      for (int j = 0; j < 64; ++j) {
        const int c = hs * 64 + j;
        buf[j] = (_Float16)((c == 0) ? sgn * t : tile[wid][row][c - 1]);
      }
      v16h* dst = (v16h*)outp;
#pragma unroll
      for (int j2 = 0; j2 < 4; ++j2) {
        v16h tv;
#pragma unroll
        for (int i = 0; i < 16; ++i) tv[i] = buf[j2 * 16 + i];
        dst[j2] = tv;
      }
    } else {
      if (hs == 0) trow[wid][row] = t;
    }
  }

  if (mat == 2) {
    // transposed store: Vt[h][d][token] so P@V B-fragments are contiguous
#pragma unroll
    for (int dd = 0; dd < 4; ++dd) {
      const int d = L * 4 + dd;   // 0..127
      v16h hv;
#pragma unroll
      for (int m = 0; m < 16; ++m) {
        const float v = (d == 0) ? trow[wid][m] : tile[wid][m][d - 1];
        hv[m] = (_Float16)v;
      }
      *(v16h*)(Vt + ((size_t)h * DMODEL + d) * TOKENS + rowbase) = hv;
    }
  }
}

// ---------------- Kernel 2: flash Lorentz attention ----------------
// 2048 wave-tasks: 8 heads x 4 batches x 64 query tiles of 16.
__global__ __launch_bounds__(128) void lmha_attn_kernel(
    const _Float16* __restrict__ Qws, const _Float16* __restrict__ Kws,
    const _Float16* __restrict__ Vt,
    const float* __restrict__ beta, const float* __restrict__ gamma,
    float* __restrict__ out, float* __restrict__ tws)
{
  __shared__ __align__(32) _Float16 Pt[4][16][64];  // per-wave prob tile

  const int wid = threadIdx.x >> 5;
  const int L   = threadIdx.x & 31;
  const int l16 = L & 15;
  const int g   = L >> 4;

  const int task  = blockIdx.x * 4 + wid;   // 0..2047
  const int h     = task / 256;
  const int rem   = task % 256;
  const int b     = rem / 64;
  const int qt    = rem % 64;
  const int qbase = b * 1024 + qt * 16;     // flat token base

  const float beta_h  = beta[h];
  const float gamma_h = gamma[h];

  // Q A-fragments (time already negated in storage)
  v16h qa[4];
  {
    const _Float16* qrow = Qws + ((size_t)h * TOKENS + qbase + l16) * DMODEL;
#pragma unroll
    for (int kt = 0; kt < 4; ++kt) qa[kt] = make_a_frag(qrow, kt, g);
  }

  v8f O[8];
#pragma unroll
  for (int dt = 0; dt < 8; ++dt) O[dt] = (v8f){0.f,0.f,0.f,0.f,0.f,0.f,0.f,0.f};
  float mrun[8], lrun[8];
#pragma unroll
  for (int r = 0; r < 8; ++r) { mrun[r] = -1e30f; lrun[r] = 0.f; }

  for (int kb = 0; kb < NSEQ; kb += 64) {
    // ---- scores: inner = <q,k>_L via plain dot with sign-folded Q ----
    v8f S[4];
#pragma unroll
    for (int nt = 0; nt < 4; ++nt) {
      v8f s = {0.f, 0.f, 0.f, 0.f, 0.f, 0.f, 0.f, 0.f};
      const _Float16* krow =
          Kws + ((size_t)h * TOKENS + b * 1024 + kb + nt * 16 + l16) * DMODEL;
#pragma unroll
      for (int kt = 0; kt < 4; ++kt) {
        v16h kf = *(const v16h*)(krow + kt * 32 + g * 16);
        s = __builtin_amdgcn_wmma_f32_16x16x32_f16(false, qa[kt], false, kf,
                                                   (short)0, s, false, false);
      }
      S[nt] = s;
    }

    // ---- arccosh distance + online softmax (per C-layout row r) ----
#pragma unroll
    for (int r = 0; r < 8; ++r) {
      float sc[4];
#pragma unroll
      for (int nt = 0; nt < 4; ++nt) {
        const float z    = fmaxf(-S[nt][r], 1.0f + 1e-6f);
        const float dist = __logf(z + __builtin_sqrtf(fmaxf(z * z - 1.0f, 0.0f)));
        sc[nt] = -beta_h * dist - gamma_h;
      }
      float bm = fmaxf(fmaxf(sc[0], sc[1]), fmaxf(sc[2], sc[3]));
#pragma unroll
      for (int m2 = 1; m2 < 16; m2 <<= 1) bm = fmaxf(bm, __shfl_xor(bm, m2, 32));
      const float mnew  = fmaxf(mrun[r], bm);
      const float alpha = __expf(mrun[r] - mnew);
      mrun[r] = mnew;
      float ps = 0.f;
#pragma unroll
      for (int nt = 0; nt < 4; ++nt) {
        const float p = __expf(sc[nt] - mnew);
        ps += p;
        Pt[wid][r + 8 * g][nt * 16 + l16] = (_Float16)p;   // C -> row-major
      }
#pragma unroll
      for (int m2 = 1; m2 < 16; m2 <<= 1) ps += __shfl_xor(ps, m2, 32);
      lrun[r] = lrun[r] * alpha + ps;
#pragma unroll
      for (int dt = 0; dt < 8; ++dt) O[dt][r] *= alpha;    // rescale accum
    }
    __builtin_amdgcn_wave_barrier();

    // ---- O += P @ V  (P: 16x64 from LDS, Vt: contiguous B-fragments) ----
    v16h pf[2];
#pragma unroll
    for (int kt = 0; kt < 2; ++kt) pf[kt] = make_a_frag(&Pt[wid][l16][0], kt, g);
#pragma unroll
    for (int dt = 0; dt < 8; ++dt) {
#pragma unroll
      for (int kt = 0; kt < 2; ++kt) {
        const _Float16* vp = Vt + ((size_t)h * DMODEL + dt * 16 + l16) * TOKENS +
                             b * 1024 + kb + kt * 32 + g * 16;
        v16h vf = *(const v16h*)vp;
        O[dt] = __builtin_amdgcn_wmma_f32_16x16x32_f16(false, pf[kt], false, vf,
                                                       (short)0, O[dt], false, false);
      }
    }
  }

  // ---- Lorentz midpoint normalization + scatter to output ----
#pragma unroll
  for (int r = 0; r < 8; ++r) {
    const float invl = 1.0f / lrun[r];
    float mid[8];
    float part = 0.f;
#pragma unroll
    for (int dt = 0; dt < 8; ++dt) {
      mid[dt] = O[dt][r] * invl;
      const int d = dt * 16 + l16;
      part += ((d == 0) ? -1.0f : 1.0f) * mid[dt] * mid[dt];
    }
#pragma unroll
    for (int m2 = 1; m2 < 16; m2 <<= 1) part += __shfl_xor(part, m2, 32);
    const float scale = 1.0f / __builtin_sqrtf(fmaxf(-part, 1e-6f));
    const int tok = qbase + r + 8 * g;
#pragma unroll
    for (int dt = 0; dt < 8; ++dt) {
      const int d = dt * 16 + l16;
      const float v = mid[dt] * scale;
      if (d == 0) tws[(size_t)h * TOKENS + tok] = v;
      else        out[(size_t)tok * OUTD + (size_t)h * 127 + d] = v;
    }
  }
}

// ---------------- Kernel 3: recombine time coordinate ----------------
__global__ void lmha_time_kernel(const float* __restrict__ tws,
                                 float* __restrict__ out)
{
  const int tok = blockIdx.x * blockDim.x + threadIdx.x;
  if (tok >= TOKENS) return;
  float s = 0.f;
#pragma unroll
  for (int h = 0; h < HEADS; ++h) { const float t = tws[h * TOKENS + tok]; s += t * t; }
  out[(size_t)tok * OUTD] = sqrtf(fmaxf(s - (float)(HEADS - 1), 1e-6f));
}

extern "C" void kernel_launch(void* const* d_in, const int* in_sizes, int n_in,
                              void* d_out, int out_size, void* d_ws, size_t ws_size,
                              hipStream_t stream) {
  (void)in_sizes; (void)n_in; (void)out_size; (void)ws_size;
  const float* x     = (const float*)d_in[0];
  const float* Wq    = (const float*)d_in[1];
  const float* bq    = (const float*)d_in[2];
  const float* Wk    = (const float*)d_in[3];
  const float* bk    = (const float*)d_in[4];
  const float* Wv    = (const float*)d_in[5];
  const float* bv    = (const float*)d_in[6];
  const float* beta  = (const float*)d_in[7];
  const float* gamma = (const float*)d_in[8];
  float* out = (float*)d_out;

  char* ws = (char*)d_ws;
  const size_t qkv_bytes = (size_t)HEADS * TOKENS * DMODEL * sizeof(_Float16); // 8 MB
  _Float16* Qws = (_Float16*)ws;
  _Float16* Kws = (_Float16*)(ws + qkv_bytes);
  _Float16* Vt  = (_Float16*)(ws + 2 * qkv_bytes);
  float*    tws = (float*)(ws + 3 * qkv_bytes);                 // 128 KB
  _Float16* Xh  = (_Float16*)(ws + 3 * qkv_bytes + (1u << 20)); // 1 MB slot after tws
  _Float16* Wh  = (_Float16*)(ws + 3 * qkv_bytes + (2u << 20)); // 768 KB

  hipLaunchKernelGGL(lmha_cvt_x_kernel, dim3(512), dim3(256), 0, stream, x, Xh);
  hipLaunchKernelGGL(lmha_cvt_w_kernel, dim3(384), dim3(256), 0, stream,
                     Wq, Wk, Wv, Wh);
  hipLaunchKernelGGL(lmha_proj_kernel, dim3(1536), dim3(128), 0, stream,
                     Xh, Wh, bq, bk, bv, Qws, Kws, Vt);
  hipLaunchKernelGGL(lmha_attn_kernel, dim3(512), dim3(128), 0, stream,
                     Qws, Kws, Vt, beta, gamma, out, tws);
  hipLaunchKernelGGL(lmha_time_kernel, dim3(16), dim3(256), 0, stream, tws, out);
}
